// EMAQuantizer_25520695673024
// MI455X (gfx1250) — matrix operand order repaired and checked by reading, not compiled, not run on hardware
//
#include <hip/hip_runtime.h>
#include <hip/hip_bf16.h>
#include <stdint.h>

typedef __attribute__((ext_vector_type(16))) __bf16 v16bf;
typedef __attribute__((ext_vector_type(8)))  __bf16 v8bf;
typedef __attribute__((ext_vector_type(8)))  float  v8f;
typedef __attribute__((ext_vector_type(4)))  int    v4i;

#define KCODES 4096
#define CDIM   64
#define HW     4096
#define BATCH  16
#define NVEC   (BATCH * HW)          /* 65536 */
#define NELEM  ((size_t)NVEC * CDIM) /* 4194304 */

#if defined(__gfx1250__) && __has_builtin(__builtin_amdgcn_global_load_async_to_lds_b128)
#define USE_ASYNC_LDS 1
#else
#define USE_ASYNC_LDS 0
#endif

#define AS1 __attribute__((address_space(1)))
#define AS3 __attribute__((address_space(3)))

// LDS staging geometry: 64 codes/stage, row padded 128B -> 144B (bank-quad spread)
#define STAGE_CODES 64
#define ROW_BYTES   144
#define STAGE_BYTES (STAGE_CODES * ROW_BYTES)   /* 9216 */

// ---------------------------------------------------------------------------
// K0a: transpose-convert inputs [B][C][HW] f32 -> x_bf [N][C] bf16
// ---------------------------------------------------------------------------
__global__ __launch_bounds__(256) void vq_prep_x(const float* __restrict__ in,
                                                 __bf16* __restrict__ x_bf) {
    __shared__ float t[64][65];
    int blk = blockIdx.x;            // 0..1023
    int b   = blk >> 6;
    int hw0 = (blk & 63) << 6;
    const float* base = in + ((size_t)b * CDIM) * HW + hw0;
    for (int i = threadIdx.x; i < 64 * 64; i += 256) {
        int c = i >> 6, s = i & 63;
        t[c][s] = base[(size_t)c * HW + s];   // coalesced over s
    }
    __syncthreads();
    __bf16* ob = x_bf + ((size_t)(b * HW + hw0)) * CDIM;
    for (int i = threadIdx.x; i < 64 * 64; i += 256) {
        int r = i >> 6, c = i & 63;
        ob[(size_t)r * CDIM + c] = (__bf16)t[c][r];  // coalesced over c
    }
}

// ---------------------------------------------------------------------------
// K0b: codebook -> bf16 copy + exact f32 row norms ||w_k||^2
// ---------------------------------------------------------------------------
__global__ __launch_bounds__(64) void vq_prep_w(const float* __restrict__ w,
                                                __bf16* __restrict__ w_bf,
                                                float* __restrict__ wnorm) {
    int k = blockIdx.x, c = threadIdx.x;
    float v = w[(size_t)k * CDIM + c];
    w_bf[(size_t)k * CDIM + c] = (__bf16)v;
    __shared__ float s[64];
    s[c] = v * v;
    __syncthreads();
    for (int off = 32; off > 0; off >>= 1) {
        if (c < off) s[c] += s[c + off];
        __syncthreads();
    }
    if (c == 0) wnorm[k] = s[0];
}

// ---------------------------------------------------------------------------
// K1: fused GEMM(-2 x.wT) + ||w||^2 + argmin via v_wmma_f32_16x16x32_bf16.
// 8 waves/block x 16 rows. Codebook streamed through LDS with double-buffered
// GLOBAL_LOAD_ASYNC_TO_LDS_B128 (ASYNCcnt), 64 codes (8KB) per stage.
// ---------------------------------------------------------------------------
__global__ __launch_bounds__(256) void vq_argmin(const __bf16* __restrict__ x_bf,
                                                 const __bf16* __restrict__ w_bf,
                                                 const float* __restrict__ wnorm,
                                                 int* __restrict__ idx_out) {
    const int lane = threadIdx.x & 31;
    const int wave = threadIdx.x >> 5;
    const int h    = lane >> 4;      // lane half
    const int m    = lane & 15;      // A-matrix row / output-col residue
    const int row0 = blockIdx.x * 128 + wave * 16;

    // A fragments for this wave's 16 rows, K-blocks [0,32) and [32,64).
    const __bf16* arow = x_bf + (size_t)(row0 + m) * CDIM;
    v8bf a0l = *(const v8bf*)(arow + 0  + 16 * h);
    v8bf a0h = *(const v8bf*)(arow + 8  + 16 * h);
    v8bf a1l = *(const v8bf*)(arow + 32 + 16 * h);
    v8bf a1h = *(const v8bf*)(arow + 40 + 16 * h);
    v16bf A0 = __builtin_shufflevector(a0l, a0h, 0,1,2,3,4,5,6,7,8,9,10,11,12,13,14,15);
    v16bf A1 = __builtin_shufflevector(a1l, a1h, 0,1,2,3,4,5,6,7,8,9,10,11,12,13,14,15);

    float    best[8];
    uint32_t bidx[8];
#pragma unroll
    for (int j = 0; j < 8; ++j) { best[j] = 3.4e38f; bidx[j] = 0u; }

#if USE_ASYNC_LDS
    __shared__ __align__(32) char lds_b[2 * STAGE_BYTES];

    // issue one stage's async copies: 256 threads x 2 B128 segs = 8KB
    auto issue_stage = [&](int s) {
        const int cbase = s * STAGE_CODES;
        const int buf   = (s & 1) * STAGE_BYTES;
#pragma unroll
        for (int r = 0; r < 2; ++r) {
            int seg = (int)threadIdx.x + r * 256;   // 0..511
            int row = seg >> 3, sg = seg & 7;       // 8 x 16B segs per 128B row
            const __bf16* gsrc = w_bf + (size_t)(cbase + row) * CDIM + sg * 8;
            char* ldst = lds_b + buf + row * ROW_BYTES + sg * 16;
            __builtin_amdgcn_global_load_async_to_lds_b128(
                (AS1 v4i*)gsrc, (AS3 v4i*)ldst, 0, 0);
        }
    };

    issue_stage(0);
    const int NSTAGES = KCODES / STAGE_CODES;       // 64
    for (int g = 0; g < NSTAGES; ++g) {
        if (g + 1 < NSTAGES) {
            issue_stage(g + 1);
#if __has_builtin(__builtin_amdgcn_s_wait_asynccnt)
            __builtin_amdgcn_s_wait_asynccnt(2);    // stage g's 2 copies done
#else
            asm volatile("s_wait_asynccnt 0x2" ::: "memory");
#endif
        } else {
#if __has_builtin(__builtin_amdgcn_s_wait_asynccnt)
            __builtin_amdgcn_s_wait_asynccnt(0);
#else
            asm volatile("s_wait_asynccnt 0x0" ::: "memory");
#endif
        }
        __syncthreads();                            // all waves' copies visible

        const char* buf = lds_b + (g & 1) * STAGE_BYTES;
#pragma unroll
        for (int c4 = 0; c4 < 4; ++c4) {
            const int code_local = c4 * 16 + m;
            const int code       = g * STAGE_CODES + code_local;
            const char* brow = buf + code_local * ROW_BYTES + 32 * h;
            v8bf b0l = *(const v8bf*)(brow + 0);
            v8bf b0h = *(const v8bf*)(brow + 16);
            v8bf b1l = *(const v8bf*)(brow + 64);
            v8bf b1h = *(const v8bf*)(brow + 80);
            v16bf B0 = __builtin_shufflevector(b0l, b0h, 0,1,2,3,4,5,6,7,8,9,10,11,12,13,14,15);
            v16bf B1 = __builtin_shufflevector(b1l, b1h, 0,1,2,3,4,5,6,7,8,9,10,11,12,13,14,15);

            v8f acc = {0.f, 0.f, 0.f, 0.f, 0.f, 0.f, 0.f, 0.f};
            acc = __builtin_amdgcn_wmma_f32_16x16x32_bf16(false, A0, false, B0,
                                                          (short)0, acc, false, false);
            acc = __builtin_amdgcn_wmma_f32_16x16x32_bf16(false, A1, false, B1,
                                                          (short)0, acc, false, false);
            const float wn = wnorm[code];
#pragma unroll
            for (int j = 0; j < 8; ++j) {
                float score = __builtin_fmaf(-2.0f, acc[j], wn);
                bool lt = score < best[j];
                best[j] = lt ? score : best[j];
                bidx[j] = lt ? (uint32_t)code : bidx[j];
            }
        }
        __syncthreads();           // done reading buf before it is re-filled
    }
#else
    // fallback: direct global B loads (WGP-scope prefetch ahead)
    for (int chunk = 0; chunk < KCODES / 16; ++chunk) {
        const int code = chunk * 16 + m;
        const __bf16* brow = w_bf + (size_t)code * CDIM + 16 * h;
        v16bf B0 = *(const v16bf*)(brow);
        v16bf B1 = *(const v16bf*)(brow + 32);
        __builtin_prefetch(brow + 16 * CDIM, 0, 3);

        v8f acc = {0.f, 0.f, 0.f, 0.f, 0.f, 0.f, 0.f, 0.f};
        acc = __builtin_amdgcn_wmma_f32_16x16x32_bf16(false, A0, false, B0,
                                                      (short)0, acc, false, false);
        acc = __builtin_amdgcn_wmma_f32_16x16x32_bf16(false, A1, false, B1,
                                                      (short)0, acc, false, false);
        const float wn = wnorm[code];
#pragma unroll
        for (int j = 0; j < 8; ++j) {
            float score = __builtin_fmaf(-2.0f, acc[j], wn);
            bool lt = score < best[j];
            best[j] = lt ? score : best[j];
            bidx[j] = lt ? (uint32_t)code : bidx[j];
        }
    }
#endif

    // reduce (min,idx) across the 16 lanes of each half (columns 0..15)
#pragma unroll
    for (int off = 1; off < 16; off <<= 1) {
#pragma unroll
        for (int j = 0; j < 8; ++j) {
            float    ov = __shfl_xor(best[j], off, 32);
            uint32_t oi = (uint32_t)__shfl_xor((int)bidx[j], off, 32);
            bool take = (ov < best[j]) || (ov == best[j] && oi < bidx[j]);
            best[j] = take ? ov : best[j];
            bidx[j] = take ? oi : bidx[j];
        }
    }
    if ((lane & 15) == 0) {
        // C/D layout: element j -> row M = j + 8*h
#pragma unroll
        for (int j = 0; j < 8; ++j) idx_out[row0 + 8 * h + j] = (int)bidx[j];
    }
}

// ---------------------------------------------------------------------------
// K2: gather quantized (output-coalesced), STE, loss partials, scatter EMA sums
// ---------------------------------------------------------------------------
__global__ __launch_bounds__(256) void vq_quant(const float* __restrict__ in,
                                                const float* __restrict__ w,
                                                const int* __restrict__ idx,
                                                float* __restrict__ q_out,
                                                float* __restrict__ idx_out_f,
                                                float* __restrict__ dw,
                                                float* __restrict__ counts,
                                                float* __restrict__ loss_part) {
    size_t e = (size_t)blockIdx.x * 256 + threadIdx.x;   // over [B][C][HW]
    int b = (int)(e >> 18);
    int c = (int)(e >> 12) & 63;
    int s = (int)e & 4095;
    int n = (b << 12) | s;
    int id = idx[n];
    float xv = in[e];
    float qv = w[(size_t)id * CDIM + c];
    q_out[e] = xv + (qv - xv);               // match reference STE rounding
    float d = qv - xv;
    atomicAdd(&dw[(size_t)id * CDIM + c], xv);
    if (c == 0) {
        atomicAdd(&counts[id], 1.0f);
        idx_out_f[n] = (float)id;
    }
    __shared__ float sm[256];
    sm[threadIdx.x] = d * d;
    __syncthreads();
    for (int off = 128; off > 0; off >>= 1) {
        if (threadIdx.x < off) sm[threadIdx.x] += sm[threadIdx.x + off];
        __syncthreads();
    }
    if (threadIdx.x == 0) loss_part[blockIdx.x] = sm[0];
}

// ---------------------------------------------------------------------------
// K3a (single block): new_cluster_size, n = sum, deterministic loss reduction
// ---------------------------------------------------------------------------
__global__ __launch_bounds__(1024) void vq_ema_a(const float* __restrict__ ecs,
                                                 const float* __restrict__ counts,
                                                 const float* __restrict__ loss_part,
                                                 float* __restrict__ ncs_out,
                                                 float* __restrict__ loss_out,
                                                 float* __restrict__ n_out) {
    __shared__ float sm[1024];
    int t = threadIdx.x;
    float acc = 0.f;
    for (int k = t; k < KCODES; k += 1024) {
        float v = 0.99f * ecs[k] + 0.01f * counts[k];
        ncs_out[k] = v;
        acc += v;
    }
    sm[t] = acc;
    __syncthreads();
    for (int off = 512; off > 0; off >>= 1) {
        if (t < off) sm[t] += sm[t + off];
        __syncthreads();
    }
    if (t == 0) n_out[0] = sm[0];
    __syncthreads();
    float lacc = 0.f;
    for (int i = t; i < 16384; i += 1024) lacc += loss_part[i];
    sm[t] = lacc;
    __syncthreads();
    for (int off = 512; off > 0; off >>= 1) {
        if (t < off) sm[t] += sm[t + off];
        __syncthreads();
    }
    if (t == 0) loss_out[0] = 0.25f * sm[0] / (float)NELEM;
}

// ---------------------------------------------------------------------------
// K3b: new_ema_w and normalized new_weight
// ---------------------------------------------------------------------------
__global__ __launch_bounds__(256) void vq_ema_b(const float* __restrict__ ema_w,
                                                const float* __restrict__ dw,
                                                const float* __restrict__ ncs,
                                                const float* __restrict__ n_ptr,
                                                float* __restrict__ new_ema_w_out,
                                                float* __restrict__ new_w_out) {
    int e = blockIdx.x * 256 + threadIdx.x;   // < 262144
    int k = e >> 6;
    float nv  = n_ptr[0];
    float nem = 0.99f * ema_w[e] + 0.01f * dw[e];
    new_ema_w_out[e] = nem;
    float cs = (ncs[k] + 1e-5f) / (nv + (float)KCODES * 1e-5f) * nv;
    new_w_out[e] = nem / cs;
}

// ---------------------------------------------------------------------------
extern "C" void kernel_launch(void* const* d_in, const int* in_sizes, int n_in,
                              void* d_out, int out_size, void* d_ws, size_t ws_size,
                              hipStream_t stream) {
    const float* inputs = (const float*)d_in[0];   // [16,64,64,64]
    const float* weight = (const float*)d_in[1];   // [4096,64]
    const float* ecs    = (const float*)d_in[2];   // [4096]
    const float* ema_w  = (const float*)d_in[3];   // [4096,64]

    // workspace layout
    char* ws = (char*)d_ws;
    size_t off = 0;
    __bf16* x_bf = (__bf16*)(ws + off); off += NELEM * 2;                 // 8 MB
    __bf16* w_bf = (__bf16*)(ws + off); off += (size_t)KCODES * CDIM * 2; // 512 KB
    float* wnorm = (float*)(ws + off);  off += (size_t)KCODES * 4;
    int*   idx   = (int*)(ws + off);    off += (size_t)NVEC * 4;
    size_t zoff  = off;                 // start of zero-init region
    float* counts    = (float*)(ws + off); off += (size_t)KCODES * 4;
    float* dw        = (float*)(ws + off); off += (size_t)KCODES * CDIM * 4;
    float* loss_part = (float*)(ws + off); off += (size_t)16384 * 4;
    size_t zbytes = off - zoff;
    float* n_scalar = (float*)(ws + off); off += 64;

    // output layout (reference return order, flattened)
    float* out       = (float*)d_out;
    float* q_out     = out;                           // 4194304
    float* loss_out  = out + NELEM;                   // 1
    float* idx_out_f = out + NELEM + 1;               // 65536
    float* new_w     = idx_out_f + NVEC;              // 262144
    float* ncs_out   = new_w + (size_t)KCODES * CDIM; // 4096
    float* new_emaw  = ncs_out + KCODES;              // 262144

    (void)hipMemsetAsync(ws + zoff, 0, zbytes, stream);
    vq_prep_x<<<1024, 256, 0, stream>>>(inputs, x_bf);
    vq_prep_w<<<KCODES, 64, 0, stream>>>(weight, w_bf, wnorm);
    vq_argmin<<<NVEC / 128, 256, 0, stream>>>(x_bf, w_bf, wnorm, idx);
    vq_quant<<<(int)(NELEM / 256), 256, 0, stream>>>(inputs, weight, idx, q_out,
                                                     idx_out_f, dw, counts, loss_part);
    vq_ema_a<<<1, 1024, 0, stream>>>(ecs, counts, loss_part, ncs_out, loss_out, n_scalar);
    vq_ema_b<<<KCODES * CDIM / 256, 256, 0, stream>>>(ema_w, dw, ncs_out, n_scalar,
                                                      new_emaw, new_w);
}
